// InvariantModel_34196529611024
// MI455X (gfx1250) — matrix operand order repaired and checked by reading, not compiled, not run on hardware
//
#include <hip/hip_runtime.h>
#include <hip/hip_bf16.h>
#include <math.h>

// ---------------- problem sizes ----------------
#define B_  32
#define L_  128
#define G_  64
#define D_  256
#define HD_ 128
#define H_  256      // 2*HD
#define N_  8
#define K_  4
#define NINP_ 128
#define NHID_ 32
#define NREL_ 16
#define E_  2048     // B*N*N
#define RH_ 512      // NREL*NHID
#define EPS_ 1e-5f

// ---------------- WMMA (CDNA5 wave32, 16x16x32 f16 -> f32) ----------------
typedef _Float16 half16 __attribute__((ext_vector_type(16)));
typedef float    float8 __attribute__((ext_vector_type(8)));
// TDM descriptor vector types (clang-23 6-arg builtin)
typedef unsigned int uint4v __attribute__((ext_vector_type(4)));
typedef int          int8v  __attribute__((ext_vector_type(8)));
typedef int          int4v  __attribute__((ext_vector_type(4)));

// A: row-major (M x K) f32 source, converted to f16 in contiguous k order (k = 16*(lane/16) + e), same as the B loaders.
__device__ __forceinline__ half16 wmma_load_a(const float* A, int lda, int m0, int k0, int lane) {
  const float* p = A + (m0 + (lane & 15)) * lda + k0 + ((lane >> 4) << 4);
  half16 r;
#pragma unroll
  for (int e = 0; e < 8; ++e)  r[e]     = (_Float16)p[e];
#pragma unroll
  for (int e = 0; e < 8; ++e)  r[e + 8] = (_Float16)p[e + 8];
  return r;
}

// B from weight stored row-major (N x K) f32 (torch Linear convention): B(k,n)=W[n][k]
__device__ __forceinline__ half16 wmma_load_b_nk(const float* W, int ldw, int n0, int k0, int lane) {
  const float* p = W + (n0 + (lane & 15)) * ldw + k0 + ((lane >> 4) << 4);
  half16 r;
#pragma unroll
  for (int e = 0; e < 16; ++e) r[e] = (_Float16)p[e];
  return r;
}

// B from f16 weight stored row-major (N x K) in LDS: 16 contiguous halves per lane.
__device__ __forceinline__ half16 wmma_load_b_nk_h(const _Float16* W, int ldw, int n0, int k0, int lane) {
  const _Float16* p = W + (n0 + (lane & 15)) * ldw + k0 + ((lane >> 4) << 4);
  half16 r;
#pragma unroll
  for (int e = 0; e < 16; ++e) r[e] = p[e];
  return r;
}

__device__ __forceinline__ void wmma_store(float* C, int ldc, int m0, int n0, int lane, float8 c) {
  int mb = m0 + ((lane >> 4) << 3);
  int n  = n0 + (lane & 15);
#pragma unroll
  for (int i = 0; i < 8; ++i) C[(mb + i) * ldc + n] = c[i];
}

// ---------------- K0: init workspace ----------------
__global__ void k_init(float* xs, int* em, float* sc) {
  int i = blockIdx.x * 256 + threadIdx.x;
  if (i < G_ * B_ * D_) xs[i] = 0.f;
  if (i < B_) em[i] = 1;
  if (i < 4) sc[i] = 0.f;
}

// ---------------- K1: embedding gather + scatter-sum into xs[g*B+b][d] ----------------
__global__ void k_embed_scatter(const int* seq, const int* pos2grp, const float* emb, float* xs) {
  int b = blockIdx.x, d = threadIdx.x;
  for (int l = 0; l < L_; ++l) {
    int g  = pos2grp[b * L_ + l];
    int tk = seq[b * L_ + l];
    xs[(g * B_ + b) * D_ + d] += emb[tk * D_ + d];
  }
}

// ---------------- K2: gate-input GEMM (both dirs): X[g*B+b][j] = xs . Wih^T + bih + bhh ----------
__global__ void k_gemm_gates(const float* xs,
                             const float* Wf, const float* Wb,
                             const float* bihf, const float* bhhf,
                             const float* bihb, const float* bhhb,
                             float* Xf, float* Xb) {
  int lane = threadIdx.x & 31;
  int tid  = blockIdx.x * 8 + (threadIdx.x >> 5);
  int dir  = tid >> 12;
  int rem  = tid & 4095;
  int m0 = (rem >> 5) * 16, n0 = (rem & 31) * 16;
  const float* W  = dir ? Wb : Wf;
  const float* bi = dir ? bihb : bihf;
  const float* bh = dir ? bhhb : bhhf;
  float* X = dir ? Xb : Xf;
  float8 c = {};
#pragma unroll
  for (int k0 = 0; k0 < 256; k0 += 32) {
    half16 a = wmma_load_a(xs, 256, m0, k0, lane);
    half16 b = wmma_load_b_nk(W, 256, n0, k0, lane);
    c = __builtin_amdgcn_wmma_f32_16x16x32_f16(false, a, false, b, (short)0, c, false, false);
  }
  float bias = bi[n0 + (lane & 15)] + bh[n0 + (lane & 15)];
#pragma unroll
  for (int i = 0; i < 8; ++i) c[i] += bias;
  wmma_store(X, 512, m0, n0, lane, c);
}

// ---------------- K3: LSTM recurrence (2 blocks = 2 directions, 1024 thr, state in LDS) -------
__global__ __launch_bounds__(1024) void k_lstm(const float* Xf, const float* Xb,
                                               const float* Whhf, const float* Whhb,
                                               float* hcat) {
  extern __shared__ float sm[];
  float* hb = sm;                    // 32*128
  float* cb = sm + B_ * HD_;         // 32*128
  float* gb = sm + 2 * B_ * HD_;     // 32*512
  int dir = blockIdx.x;
  const float* X   = dir ? Xb : Xf;
  const float* Whh = dir ? Whhb : Whhf;
  int t = threadIdx.x;
  for (int i = t; i < 2 * B_ * HD_; i += 1024) sm[i] = 0.f;
  __syncthreads();
  int b  = t >> 5;
  int j0 = t & 31;
  for (int s = 0; s < G_; ++s) {
    int g = dir ? (G_ - 1 - s) : s;
    const float* Xrow = X + (g * B_ + b) * 512;
    float acc[16];
#pragma unroll
    for (int i = 0; i < 16; ++i) acc[i] = Xrow[j0 + 32 * i];
    for (int k = 0; k < HD_; ++k) {
      float hk = hb[b * HD_ + k];
#pragma unroll
      for (int i = 0; i < 16; ++i) acc[i] += hk * Whh[(j0 + 32 * i) * HD_ + k];
    }
#pragma unroll
    for (int i = 0; i < 16; ++i) gb[b * 512 + j0 + 32 * i] = acc[i];
    __syncthreads();
#pragma unroll
    for (int q = 0; q < 4; ++q) {
      int idx = t + q * 1024;
      int bb = idx >> 7, k = idx & 127;
      float gi = gb[bb * 512 + k];
      float gf = gb[bb * 512 + 128 + k];
      float gg = gb[bb * 512 + 256 + k];
      float go = gb[bb * 512 + 384 + k];
      float si = 1.f / (1.f + __expf(-gi));
      float sf = 1.f / (1.f + __expf(-gf));
      float so = 1.f / (1.f + __expf(-go));
      float cv = sf * cb[bb * HD_ + k] + si * tanhf(gg);
      float hv = so * tanhf(cv);
      cb[bb * HD_ + k] = cv;
      hb[bb * HD_ + k] = hv;
      hcat[(bb * G_ + g) * H_ + dir * HD_ + k] = hv;
    }
    __syncthreads();
  }
}

// ---------------- K4: node gather + sum over K -> h_node(256,256) ----------------
__global__ void k_gather_nodes(const int* pos2grp, const int* idxp, const float* hcat, float* hn) {
  int node = blockIdx.x, col = threadIdx.x;
  int b = node >> 3, n = node & 7;
  float acc = 0.f;
#pragma unroll
  for (int kk = 0; kk < K_; ++kk) {
    int l = idxp[(b * N_ + n) * K_ + kk];
    int g = pos2grp[b * L_ + l];
    acc += hcat[(b * G_ + g) * H_ + col];
  }
  hn[node * H_ + col] = acc;
}

// ---------------- K5: endpoint projections (WMMA GEMM, M=256 N=128 K=256, 2 mats) -------------
__global__ void k_gemm_proj(const float* hn,
                            const float* Ws, const float* bs,
                            const float* Wd, const float* bd,
                            float* P0, float* P1) {
  int lane = threadIdx.x & 31;
  int tid  = blockIdx.x * 8 + (threadIdx.x >> 5);
  int mat  = tid >> 7, rem = tid & 127;
  int m0 = (rem >> 3) * 16, n0 = (rem & 7) * 16;
  const float* W    = mat ? Wd : Ws;
  const float* bias = mat ? bd : bs;
  float* P = mat ? P1 : P0;
  float8 c = {};
#pragma unroll
  for (int k0 = 0; k0 < 256; k0 += 32) {
    half16 a = wmma_load_a(hn, 256, m0, k0, lane);
    half16 b = wmma_load_b_nk(W, 256, n0, k0, lane);
    c = __builtin_amdgcn_wmma_f32_16x16x32_f16(false, a, false, b, (short)0, c, false, false);
  }
  float bv = bias[n0 + (lane & 15)];
#pragma unroll
  for (int i = 0; i < 8; ++i) c[i] += bv;
  wmma_store(P, NINP_, m0, n0, lane, c);
}

// ---------------- K6: BatchNorm over the 256 node rows ----------------
__global__ void k_bn_cols(float* P0, float* P1,
                          const float* g0, const float* b0,
                          const float* g1, const float* b1) {
  __shared__ float r1[256], r2[256];
  int mat = blockIdx.x >> 7, col = blockIdx.x & 127;
  float* P = mat ? P1 : P0;
  const float* gp = mat ? g1 : g0;
  const float* bp = mat ? b1 : b0;
  int t = threadIdx.x;
  float x = P[t * NINP_ + col];
  r1[t] = x; r2[t] = x * x;
  __syncthreads();
  for (int s = 128; s > 0; s >>= 1) {
    if (t < s) { r1[t] += r1[t + s]; r2[t] += r2[t + s]; }
    __syncthreads();
  }
  float m   = r1[0] / 256.f;
  float var = r2[0] / 256.f - m * m;
  float inv = rsqrtf(var + EPS_);
  P[t * NINP_ + col] = (x - m) * inv * gp[col] + bp[col];
}

// ---------------- K7: NTL per (r,h) ----------------
// Stage W[r,h] (128x128 f32, 64KB) into LDS via Tensor Data Mover (one wave issues,
// TENSORcnt-tracked), transpose-convert to f16 (N,K) layout, then T = P0 @ W with WMMA
// (B fragments loaded once per wave from LDS, reused across 16 m-tiles), then per-edge dots.
__global__ void k_ntl_edges(const float* P0, const float* P1,
                            const float* ntlw, const float* ntlv, const float* ntlb,
                            const int* u, const int* v, float* pre) {
  extern __shared__ float sm[];
  float*     T    = sm;                          // 32768 f32 = 128 KB
  float*     Wf32 = sm + 32768;                  // 16384 f32 =  64 KB (TDM dest)
  _Float16*  Wt   = (_Float16*)(sm + 49152);     // 16384 f16 =  32 KB, (N,K) layout
  float*     Vc   = sm + 57344;                  // 256 f32
  int rh   = blockIdx.x;
  int lane = threadIdx.x & 31;
  int w    = threadIdx.x >> 5;

  // --- TDM: DMA 2D tile (128x128 f32) from ntl_w + rh*16384 into LDS ---
  if (threadIdx.x < 32) {
    unsigned lds_addr = (unsigned)(size_t)(void*)Wf32;        // flat addr low 32 bits = LDS offset
    unsigned long long ga = (unsigned long long)(size_t)(const void*)(ntlw + rh * NINP_ * NINP_);
    uint4v g0;
    g0[0] = 1u;                                               // count=1, user mode, no gather
    g0[1] = lds_addr;                                         // lds_addr[31:0]
    g0[2] = (unsigned)(ga & 0xFFFFFFFFu);                     // global_addr[31:0]
    g0[3] = (unsigned)((ga >> 32) & 0x1FFFFFFu) | (2u << 30); // global_addr[56:32] | type=2
    int8v g1;
    g1[0] = (2 << 16);          // workgroup_mask=0, data_size=2 (4 bytes)
    g1[1] = (128 << 16);        // tensor_dim0[15:0] in bits 63:48
    g1[2] = (128 << 16);        // tensor_dim0 hi=0 | tensor_dim1[15:0]
    g1[3] = (128 << 16);        // tensor_dim1 hi=0 | tile_dim0=128
    g1[4] = 128;                // tile_dim1=128, tile_dim2=0
    g1[5] = 128;                // tensor_dim0_stride low32
    g1[6] = 0;                  // stride0 hi | stride1 lo
    g1[7] = 0;                  // stride1 hi
    int4v g2 = {0, 0, 0, 0};
    int4v g3 = {0, 0, 0, 0};
    int8v g4 = {0, 0, 0, 0, 0, 0, 0, 0};
    __builtin_amdgcn_tensor_load_to_lds(g0, g1, g2, g3, g4, 0);
    __builtin_amdgcn_s_wait_tensorcnt(0);
  }
  Vc[threadIdx.x] = ntlv[rh * 2 * NINP_ + threadIdx.x];
  __syncthreads();

  // --- transpose-convert W (K,N) f32 -> Wt (N,K) f16 in LDS ---
  for (int idx = threadIdx.x; idx < NINP_ * NINP_; idx += 256) {
    int k = idx >> 7, n = idx & 127;
    Wt[n * NINP_ + k] = (_Float16)Wf32[k * NINP_ + n];
  }
  __syncthreads();

  // --- phase A: T = P0 @ W, wave w owns n-tile w; B fragments loaded once ---
  int n0 = w * 16;
  half16 bfrag[4];
#pragma unroll
  for (int kk = 0; kk < 4; ++kk)
    bfrag[kk] = wmma_load_b_nk_h(Wt, NINP_, n0, kk * 32, lane);
#pragma unroll 1
  for (int mt = 0; mt < 16; ++mt) {
    int m0 = mt * 16;
    float8 c = {};
#pragma unroll
    for (int kk = 0; kk < 4; ++kk) {
      half16 a = wmma_load_a(P0, NINP_, m0, kk * 32, lane);
      c = __builtin_amdgcn_wmma_f32_16x16x32_f16(false, a, false, bfrag[kk], (short)0, c, false, false);
    }
    wmma_store(T, NINP_, m0, n0, lane, c);
  }
  __syncthreads();

  // --- phase B: per-edge bilinear + linear + bias ---
  float bias = ntlb[rh];
  for (int e = threadIdx.x; e < E_; e += 256) {
    int un = u[e], vn = v[e];
    const float* p0 = P0 + un * NINP_;
    const float* p1 = P1 + vn * NINP_;
    const float* Tr = T  + un * NINP_;
    float acc = bias;
    for (int j = 0; j < NINP_; ++j) {
      float q = p1[j];
      acc += Tr[j] * q + Vc[j] * p0[j] + Vc[NINP_ + j] * q;
    }
    pre[rh * E_ + e] = acc;
  }
}

// ---------------- K8: BN over E per (r,h) slice + tanh (in place) ----------------
__global__ void k_bn_tanh(float* pre, const float* g, const float* b) {
  __shared__ float r1[256], r2[256];
  int rh = blockIdx.x, t = threadIdx.x;
  float* row = pre + rh * E_;
  float s1 = 0.f, s2 = 0.f, xv[8];
#pragma unroll
  for (int q = 0; q < 8; ++q) { float x = row[t + q * 256]; xv[q] = x; s1 += x; s2 += x * x; }
  r1[t] = s1; r2[t] = s2;
  __syncthreads();
  for (int s = 128; s > 0; s >>= 1) {
    if (t < s) { r1[t] += r1[t + s]; r2[t] += r2[t + s]; }
    __syncthreads();
  }
  float m   = r1[0] / (float)E_;
  float var = r2[0] / (float)E_ - m * m;
  float inv = rsqrtf(var + EPS_);
  float gg = g[rh], bb = b[rh];
#pragma unroll
  for (int q = 0; q < 8; ++q) row[t + q * 256] = tanhf((xv[q] - m) * inv * gg + bb);
}

// ---------------- K9: logits[e][r] = sum_h u[r][h] * act[r][h][e] ----------------
__global__ void k_logit(const float* pre, const float* U, float* out) {
  int o = blockIdx.x * 256 + threadIdx.x;
  int e = o >> 4, r = o & 15;
  float acc = 0.f;
#pragma unroll
  for (int h = 0; h < NHID_; ++h)
    acc += U[r * NHID_ + h] * pre[(r * NHID_ + h) * E_ + e];
  out[o] = acc;
}

// ---------------- K10: metrics ----------------
__global__ void k_metrics(const float* logit, const unsigned char* mask, int* em, float* sc) {
  int e = blockIdx.x * 256 + threadIdx.x;
  if (e >= E_) return;
  int b = e >> 6;
  int cnt = 0, all = 1;
  float lp = 0.f, ln = 0.f;
#pragma unroll
  for (int r = 0; r < NREL_; ++r) {
    float x  = logit[e * NREL_ + r];
    bool msk = mask[e * NREL_ + r] != 0;
    bool eq  = ((x > 0.f) == msk);
    cnt += (int)eq; all &= (int)eq;
    if (msk) {
      float ls = (x >= 0.f) ? -log1pf(__expf(-x)) : (x - log1pf(__expf(x)));
      lp -= ls;
    } else {
      float sig = 1.f / (1.f + __expf(-x));
      ln -= logf(1.0f + 1e-5f - sig);
    }
  }
  if (!all) atomicAnd(em + b, 0);
  atomicAdd(sc + 0, (float)cnt);
  atomicAdd(sc + 1, lp);
  atomicAdd(sc + 2, ln);
}

__global__ void k_finalize(const int* em, const float* sc, float* out) {
  if (threadIdx.x == 0 && blockIdx.x == 0) {
    float ems = 0.f;
    for (int i = 0; i < B_; ++i) ems += (float)em[i];
    out[E_ * NREL_ + 0] = (sc[1] + sc[2]) / (float)B_;
    out[E_ * NREL_ + 1] = sc[0] / (float)(E_ * NREL_);
    out[E_ * NREL_ + 2] = ems / (float)B_;
  }
}

// ---------------- launch ----------------
extern "C" void kernel_launch(void* const* d_in, const int* in_sizes, int n_in,
                              void* d_out, int out_size, void* d_ws, size_t ws_size,
                              hipStream_t stream) {
  (void)in_sizes; (void)n_in; (void)out_size; (void)ws_size;
  const int*   seq      = (const int*)d_in[0];
  const int*   pos2grp  = (const int*)d_in[1];
  const int*   idxp     = (const int*)d_in[2];
  const int*   u        = (const int*)d_in[3];
  const int*   v        = (const int*)d_in[4];
  const unsigned char* mask = (const unsigned char*)d_in[5];
  const float* emb      = (const float*)d_in[6];
  const float* Wih_f    = (const float*)d_in[7];
  const float* Whh_f    = (const float*)d_in[8];
  const float* bih_f    = (const float*)d_in[9];
  const float* bhh_f    = (const float*)d_in[10];
  const float* Wih_b    = (const float*)d_in[11];
  const float* Whh_b    = (const float*)d_in[12];
  const float* bih_b    = (const float*)d_in[13];
  const float* bhh_b    = (const float*)d_in[14];
  const float* lin_src_w = (const float*)d_in[15];
  const float* lin_src_b = (const float*)d_in[16];
  const float* lin_dst_w = (const float*)d_in[17];
  const float* lin_dst_b = (const float*)d_in[18];
  const float* bn_src_g = (const float*)d_in[19];
  const float* bn_src_b = (const float*)d_in[20];
  const float* bn_dst_g = (const float*)d_in[21];
  const float* bn_dst_b = (const float*)d_in[22];
  const float* ntl_w    = (const float*)d_in[23];
  const float* ntl_v    = (const float*)d_in[24];
  const float* ntl_b    = (const float*)d_in[25];
  const float* ntl_u    = (const float*)d_in[26];
  const float* ntl_bn_g = (const float*)d_in[27];
  const float* ntl_bn_b = (const float*)d_in[28];

  float* ws = (float*)d_ws;
  float* xs   = ws;                       // 524288  (G*B, D)
  float* Xf   = ws + 524288;              // 1048576 (G*B, 512)
  float* Xb   = ws + 1572864;             // 1048576
  float* hcat = ws + 2621440;             // 524288  (B, G, H)
  float* hn   = ws + 3145728;             // 65536   (256, 256)
  float* P0   = ws + 3211264;             // 32768   (256, 128)
  float* P1   = ws + 3244032;             // 32768
  float* pre  = ws + 3276800;             // 1048576 (512, 2048)
  int*   em   = (int*)(ws + 4325376);     // 32
  float* sc   = ws + 4325440;             // 4

  float* out = (float*)d_out;

  k_init<<<2048, 256, 0, stream>>>(xs, em, sc);
  k_embed_scatter<<<B_, 256, 0, stream>>>(seq, pos2grp, emb, xs);
  k_gemm_gates<<<1024, 256, 0, stream>>>(xs, Wih_f, Wih_b, bih_f, bhh_f, bih_b, bhh_b, Xf, Xb);
  k_lstm<<<2, 1024, (2 * B_ * HD_ + B_ * 512) * sizeof(float), stream>>>(Xf, Xb, Whh_f, Whh_b, hcat);
  k_gather_nodes<<<B_ * N_, 256, 0, stream>>>(pos2grp, idxp, hcat, hn);
  k_gemm_proj<<<32, 256, 0, stream>>>(hn, lin_src_w, lin_src_b, lin_dst_w, lin_dst_b, P0, P1);
  k_bn_cols<<<256, 256, 0, stream>>>(P0, P1, bn_src_g, bn_src_b, bn_dst_g, bn_dst_b);
  // LDS: T 128KB + Wf32 64KB + Wt(f16) 32KB + Vc 1KB = 230400 bytes (WGP has 320KB)
  k_ntl_edges<<<RH_, 256, 230400, stream>>>(P0, P1, ntl_w, ntl_v, ntl_b, u, v, pre);
  k_bn_tanh<<<RH_, 256, 0, stream>>>(pre, ntl_bn_g, ntl_bn_b);
  k_logit<<<128, 256, 0, stream>>>(pre, ntl_u, out);
  k_metrics<<<8, 256, 0, stream>>>(out, mask, em, sc);
  k_finalize<<<1, 32, 0, stream>>>(em, sc, out);
}
